// fromScratchLSTM_5514738008108
// MI455X (gfx1250) — matrix operand order repaired and checked by reading, not compiled, not run on hardware
//
#include <hip/hip_runtime.h>
#include <hip/hip_bf16.h>
#include <math.h>

// ---------------------------------------------------------------------------
// LSTM on MI455X (gfx1250): fused per-step [x_t | h] @ [Wx;Wh] GEMM in bf16
// WMMA with fp32 accumulation, gate nonlinearities fused via LDS, and the
// h-state staged into LDS with CDNA5 async global->LDS copies overlapped
// with the x-phase of the GEMM.
// ---------------------------------------------------------------------------

typedef __attribute__((ext_vector_type(16))) __bf16 v16bf;
typedef __attribute__((ext_vector_type(8)))  __bf16 v8bf;
typedef __attribute__((ext_vector_type(8)))  float  v8f;
typedef __attribute__((ext_vector_type(4)))  float  v4f;
typedef __attribute__((ext_vector_type(4)))  int    v4i;

#define I_DIM 1024
#define H_DIM 1024
#define B_DIM 64
#define S_DIM 512
#define K_TOT 2048          // I + H
#define KT_TOT 64           // 2048 / 32 k tiles

#if __has_builtin(__builtin_amdgcn_global_load_async_to_lds_b128)
#define HAVE_ASYNC_LDS 1
// exact builtin parameter types: (int4 AS1*, int4 AS3*, imm, imm)
typedef __attribute__((address_space(1))) v4i GLB_v4i;
typedef __attribute__((address_space(3))) v4i LDS_v4i;
#endif

// LDS layout (dynamic): [ h tile 128KB | gates 16KB ]
#define LDS_H_BYTES   (B_DIM * H_DIM * 2)
#define LDS_BYTES     (LDS_H_BYTES + 4 * B_DIM * 16 * 4)

// ---------------------------------------------------------------------------
// Pack [Wx;Wh] (fp32, four gates each) into bf16 WMMA B-fragment layout:
//   Wpack[((nt*KT_TOT + kt)*32 + lane)*16 + e] =
//     W[kt*32 + (lane>>4)*8 + (e&7) + (e>>3)*16][nt*16 + (lane&15)]
// (lane%16 indexes N, element indexes K in table order) so the step
// kernel's B load is one contiguous 32B read per lane.
// ---------------------------------------------------------------------------
__global__ void pack_weights_kernel(const float* __restrict__ Wi_i, const float* __restrict__ Wh_i,
                                    const float* __restrict__ Wi_f, const float* __restrict__ Wh_f,
                                    const float* __restrict__ Wi_g, const float* __restrict__ Wh_g,
                                    const float* __restrict__ Wi_o, const float* __restrict__ Wh_o,
                                    __bf16* __restrict__ Wpack) {
    size_t o = (size_t)blockIdx.x * blockDim.x + threadIdx.x;   // < 2048*4096
    int e    = (int)(o & 15);
    int lane = (int)((o >> 4) & 31);
    int kt   = (int)((o >> 9) & 63);
    int nt   = (int)(o >> 15);               // 0..255
    int k    = kt * 32 + (lane >> 4) * 8 + (e & 7) + (e >> 3) * 16;   // 0..2047  (0..1023 = Wx rows)
    int n    = nt * 16 + (lane & 15);        // 0..4095
    int gate = n >> 10;
    int j    = n & 1023;
    float v;
    if (k < I_DIM) {
        const float* W = (gate == 0) ? Wi_i : (gate == 1) ? Wi_f : (gate == 2) ? Wi_g : Wi_o;
        v = W[(size_t)k * H_DIM + j];
    } else {
        const float* W = (gate == 0) ? Wh_i : (gate == 1) ? Wh_f : (gate == 2) ? Wh_g : Wh_o;
        v = W[(size_t)(k - I_DIM) * H_DIM + j];
    }
    Wpack[o] = (__bf16)v;
}

__global__ void init_state_kernel(float* __restrict__ c, __bf16* __restrict__ h0) {
    int idx = blockIdx.x * blockDim.x + threadIdx.x;   // < 65536
    if (idx < B_DIM * H_DIM) {
        c[idx]  = 0.0f;
        h0[idx] = (__bf16)0.0f;
    }
}

__device__ __forceinline__ float sigmoidf_fast(float x) {
    return 1.0f / (1.0f + __expf(-x));
}

// A fragment (16x32 bf16) from fp32 x row: lane = M, elems 0..7 -> K=base..+7,
// elems 8..15 -> K=base+16..+23, base = kt*32 + lanehi*8.
__device__ __forceinline__ v16bf load_a_x(const float* __restrict__ xrow, int kt, int lanehi) {
    const float* p = xrow + kt * 32 + lanehi * 8;
    v4f f0 = *(const v4f*)(p);
    v4f f1 = *(const v4f*)(p + 4);
    v4f f2 = *(const v4f*)(p + 16);
    v4f f3 = *(const v4f*)(p + 20);
    v16bf a;
    #pragma unroll
    for (int q = 0; q < 4; ++q) {
        a[q]      = (__bf16)f0[q];
        a[q + 4]  = (__bf16)f1[q];
        a[q + 8]  = (__bf16)f2[q];
        a[q + 12] = (__bf16)f3[q];
    }
    return a;
}

// A fragment from bf16 LDS h row (kt counts from 0 within the h phase).
__device__ __forceinline__ v16bf load_a_h(const __bf16* __restrict__ hrow, int kt, int lanehi) {
    const __bf16* p = hrow + kt * 32 + lanehi * 8;
    v8bf g0 = *(const v8bf*)(p);
    v8bf g1 = *(const v8bf*)(p + 16);
    v16bf a;
    #pragma unroll
    for (int q = 0; q < 8; ++q) { a[q] = g0[q]; a[q + 8] = g1[q]; }
    return a;
}

#define WMMA_BF16(A, B, C) \
    __builtin_amdgcn_wmma_f32_16x16x32_bf16(false, (A), false, (B), (short)0, (C), false, false)

// ---------------------------------------------------------------------------
// One LSTM timestep.
// grid = 64 WGs (jt = 16-column strip of H), block = 256 threads (8 waves).
// Wave w: M-tile mt = w%4, gate pair {2*(w/4), 2*(w/4)+1} -> shares A fragment.
//   1) kick off async copy h_{t-1} (128 KB bf16) global->LDS  (ASYNCcnt)
//   2) x-phase GEMM (K=0..1023) from global x, overlapped with (1)
//   3) s_wait_asynccnt 0 + barrier
//   4) h-phase GEMM (K=1024..2047) with A fragments from LDS (ds_load_b128)
//   5) gates -> LDS, fused sigmoid/tanh cell update
// ---------------------------------------------------------------------------
__global__ __launch_bounds__(256)
void lstm_step_kernel(const float*  __restrict__ x,       // (B,S,I) fp32
                      const __bf16* __restrict__ Wpack,   // packed [Wx;Wh] bf16
                      const float*  __restrict__ b_i, const float* __restrict__ b_f,
                      const float*  __restrict__ b_g, const float* __restrict__ b_o,
                      float*        __restrict__ c,       // (B,H) fp32 state
                      const __bf16* __restrict__ h_in,    // (B,H) bf16 state (read)
                      __bf16*       __restrict__ h_out,   // (B,H) bf16 state (write)
                      float*        __restrict__ out,     // hidden seq (B,S,H) ++ hT ++ cT
                      int t) {
    extern __shared__ __align__(16) char smem[];
    __bf16* lds_h     = (__bf16*)smem;                    // 64 x 1024 bf16
    float*  lds_gates = (float*)(smem + LDS_H_BYTES);     // [gate][m][jl]

    const int tid    = threadIdx.x;
    const int wave   = tid >> 5;
    const int lane   = tid & 31;
    const int lanehi = lane >> 4;
    const int lane15 = lane & 15;
    const int jt     = blockIdx.x;           // 0..63
    const int j0     = jt * 16;

    // ---- (1) stage h_{t-1} into LDS: 8192 chunks of 16B, 32 per thread -----
#ifdef HAVE_ASYNC_LDS
    #pragma unroll
    for (int it = 0; it < 32; ++it) {
        const int idx = it * 256 + tid;      // 16B chunk id
        __builtin_amdgcn_global_load_async_to_lds_b128(
            (GLB_v4i*)(h_in + (size_t)idx * 8), (LDS_v4i*)(lds_h + (size_t)idx * 8), 0, 0);
    }
#else
    #pragma unroll
    for (int it = 0; it < 32; ++it) {
        const int idx = it * 256 + tid;
        *(v8bf*)(lds_h + (size_t)idx * 8) = *(const v8bf*)(h_in + (size_t)idx * 8);
    }
#endif

    const int mt = wave & 3;                 // M tile (batch rows mt*16..mt*16+15)
    const int g0 = (wave >> 2) * 2;          // gates g0 and g0+1
    const int m  = mt * 16 + lane15;         // this lane's A-matrix row (batch idx)

    const float* bp0 = (g0 == 0) ? b_i : b_g;
    const float* bp1 = (g0 == 0) ? b_f : b_o;
    const float bv0 = bp0[j0 + lane15];
    const float bv1 = bp1[j0 + lane15];
    v8f acc0a = {bv0, bv0, bv0, bv0, bv0, bv0, bv0, bv0};
    v8f acc1a = {bv1, bv1, bv1, bv1, bv1, bv1, bv1, bv1};
    v8f acc0b = {};
    v8f acc1b = {};

    const int nt0 = g0 * 64 + jt;            // column tile for gate g0
    const int nt1 = nt0 + 64;                // column tile for gate g0+1
    const __bf16* wp0  = Wpack + ((size_t)nt0 * KT_TOT) * 32 * 16 + (size_t)lane * 16;
    const __bf16* wp1  = Wpack + ((size_t)nt1 * KT_TOT) * 32 * 16 + (size_t)lane * 16;
    const float*  xrow = x + ((size_t)m * S_DIM + t) * I_DIM;

    // ---- (2) x-phase: K = 0..1023, 4 independent WMMA chains ---------------
    for (int kt = 0; kt < 32; kt += 2) {
        v16bf a0  = load_a_x(xrow, kt, lanehi);
        v16bf a1  = load_a_x(xrow, kt + 1, lanehi);
        v16bf b00 = *(const v16bf*)(wp0 + (size_t)kt * 512);
        v16bf b01 = *(const v16bf*)(wp1 + (size_t)kt * 512);
        v16bf b10 = *(const v16bf*)(wp0 + (size_t)(kt + 1) * 512);
        v16bf b11 = *(const v16bf*)(wp1 + (size_t)(kt + 1) * 512);
        acc0a = WMMA_BF16(a0, b00, acc0a);
        acc1a = WMMA_BF16(a0, b01, acc1a);
        acc0b = WMMA_BF16(a1, b10, acc0b);
        acc1b = WMMA_BF16(a1, b11, acc1b);
    }

    // ---- (3) make staged h visible -----------------------------------------
#ifdef HAVE_ASYNC_LDS
#if __has_builtin(__builtin_amdgcn_s_wait_asynccnt)
    __builtin_amdgcn_s_wait_asynccnt(0);
#else
    asm volatile("s_wait_asynccnt 0x0" ::: "memory");
#endif
#endif
    __syncthreads();

    // ---- (4) h-phase: K = 1024..2047, A fragments from LDS -----------------
    {
        const __bf16* hrow = lds_h + (size_t)m * H_DIM;
        const __bf16* wph0 = wp0 + (size_t)32 * 512;
        const __bf16* wph1 = wp1 + (size_t)32 * 512;
        for (int kt = 0; kt < 32; kt += 2) {
            v16bf a0  = load_a_h(hrow, kt, lanehi);
            v16bf a1  = load_a_h(hrow, kt + 1, lanehi);
            v16bf b00 = *(const v16bf*)(wph0 + (size_t)kt * 512);
            v16bf b01 = *(const v16bf*)(wph1 + (size_t)kt * 512);
            v16bf b10 = *(const v16bf*)(wph0 + (size_t)(kt + 1) * 512);
            v16bf b11 = *(const v16bf*)(wph1 + (size_t)(kt + 1) * 512);
            acc0a = WMMA_BF16(a0, b00, acc0a);
            acc1a = WMMA_BF16(a0, b01, acc1a);
            acc0b = WMMA_BF16(a1, b10, acc0b);
            acc1b = WMMA_BF16(a1, b11, acc1b);
        }
    }
    const v8f acc0 = acc0a + acc0b;
    const v8f acc1 = acc1a + acc1b;

    // ---- (5) spill gate tiles to LDS: VGPR r -> M = r + 8*lanehi, N = lane15
    __syncthreads();   // lds_h no longer needed; gates region reused safely
    #pragma unroll
    for (int r = 0; r < 8; ++r) {
        const int mloc = mt * 16 + r + 8 * lanehi;  // 0..63
        lds_gates[(g0 * B_DIM + mloc) * 16 + lane15]       = acc0[r];
        lds_gates[((g0 + 1) * B_DIM + mloc) * 16 + lane15] = acc1[r];
    }
    __syncthreads();

    // ---- fused LSTM cell update: 1024 elements, 4 per thread ---------------
    #pragma unroll
    for (int it = 0; it < 4; ++it) {
        const int idx = it * 256 + tid;     // 0..1023
        const int mm  = idx >> 4;           // batch row 0..63
        const int jl  = idx & 15;
        const float gi = lds_gates[(0 * B_DIM + mm) * 16 + jl];
        const float gf = lds_gates[(1 * B_DIM + mm) * 16 + jl];
        const float gg = lds_gates[(2 * B_DIM + mm) * 16 + jl];
        const float go = lds_gates[(3 * B_DIM + mm) * 16 + jl];
        const float iv = sigmoidf_fast(gi);
        const float fv = sigmoidf_fast(gf);
        const float gv = tanhf(gg);
        const float ov = sigmoidf_fast(go);
        const size_t cidx = (size_t)mm * H_DIM + j0 + jl;
        const float cv = fv * c[cidx] + iv * gv;
        const float hv = ov * tanhf(cv);
        c[cidx]     = cv;
        h_out[cidx] = (__bf16)hv;
        out[((size_t)mm * S_DIM + t) * H_DIM + j0 + jl] = hv;
        if (t == S_DIM - 1) {
            const size_t hT_off = (size_t)B_DIM * S_DIM * H_DIM;
            out[hT_off + cidx]                         = hv;   // h_T
            out[hT_off + (size_t)B_DIM * H_DIM + cidx] = cv;   // c_T
        }
    }
}

extern "C" void kernel_launch(void* const* d_in, const int* in_sizes, int n_in,
                              void* d_out, int out_size, void* d_ws, size_t ws_size,
                              hipStream_t stream) {
    (void)in_sizes; (void)n_in; (void)out_size; (void)ws_size;

    const float* x    = (const float*)d_in[0];
    const float* Wi_i = (const float*)d_in[1];
    const float* Wh_i = (const float*)d_in[2];
    const float* b_i  = (const float*)d_in[3];
    const float* Wi_f = (const float*)d_in[4];
    const float* Wh_f = (const float*)d_in[5];
    const float* b_f  = (const float*)d_in[6];
    const float* Wi_g = (const float*)d_in[7];
    const float* Wh_g = (const float*)d_in[8];
    const float* b_g  = (const float*)d_in[9];
    const float* Wi_o = (const float*)d_in[10];
    const float* Wh_o = (const float*)d_in[11];
    const float* b_o  = (const float*)d_in[12];

    // workspace layout
    char* ws = (char*)d_ws;
    __bf16* Wpack = (__bf16*)ws;                                 // 2048*4096 bf16 = 16 MB
    ws += (size_t)K_TOT * 4 * H_DIM * sizeof(__bf16);
    float* c = (float*)ws;                                       // 64*1024 fp32 = 256 KB
    ws += (size_t)B_DIM * H_DIM * sizeof(float);
    __bf16* hbuf0 = (__bf16*)ws;                                 // 128 KB
    ws += (size_t)B_DIM * H_DIM * sizeof(__bf16);
    __bf16* hbuf1 = (__bf16*)ws;                                 // 128 KB

    // 1) pack weights -> bf16 WMMA fragment layout (L2-resident for the scan)
    {
        const size_t total = (size_t)K_TOT * 4 * H_DIM;          // 8.4M elements
        pack_weights_kernel<<<(unsigned)(total / 256), 256, 0, stream>>>(
            Wi_i, Wh_i, Wi_f, Wh_f, Wi_g, Wh_g, Wi_o, Wh_o, Wpack);
    }
    // 2) zero h0 / c0
    init_state_kernel<<<(B_DIM * H_DIM + 255) / 256, 256, 0, stream>>>(c, hbuf0);

    // 3) sequential scan: one fused GEMM+cell kernel per timestep
    float* out = (float*)d_out;
    for (int t = 0; t < S_DIM; ++t) {
        const __bf16* h_in  = (t & 1) ? hbuf1 : hbuf0;
        __bf16*       h_out = (t & 1) ? hbuf0 : hbuf1;
        lstm_step_kernel<<<H_DIM / 16, 256, LDS_BYTES, stream>>>(
            x, Wpack, b_i, b_f, b_g, b_o, c, h_in, h_out, out, t);
    }
}
